// ConditionedMADE_91225105367640
// MI455X (gfx1250) — matrix operand order repaired and checked by reading, not compiled, not run on hardware
//
#include <hip/hip_runtime.h>
#include <hip/hip_bf16.h>

// ---------------------------------------------------------------------------
// Types for CDNA5 WMMA (wave32, 16x16x32 bf16 -> f32)
// ---------------------------------------------------------------------------
typedef __attribute__((ext_vector_type(16))) __bf16 v16bf;
typedef __attribute__((ext_vector_type(8)))  float  v8f;

#define BATCH   2048
#define NIN     16
#define NH      1024
#define NCOND   128

// ---------------------------------------------------------------------------
// bf16 helpers (round-to-nearest-even, bit ops only)
// ---------------------------------------------------------------------------
__device__ __forceinline__ unsigned short f2bf(float f) {
    unsigned int u = __float_as_uint(f);
    u += 0x7FFFu + ((u >> 16) & 1u);
    return (unsigned short)(u >> 16);
}
__device__ __forceinline__ float bf2f(unsigned short h) {
    return __uint_as_float(((unsigned int)h) << 16);
}

// ---------------------------------------------------------------------------
// WMMA wrapper
// ---------------------------------------------------------------------------
__device__ __forceinline__ v8f wmma_bf16(v16bf a, v16bf b, v8f c) {
    return __builtin_amdgcn_wmma_f32_16x16x32_bf16(
        /*neg_a=*/false, a, /*neg_b=*/false, b,
        /*c_mod=*/(short)0, c, /*reuse_a=*/false, /*reuse_b=*/false);
}

// A-matrix 16x32 bf16 lane layout: lane m=L%16; lanes<16 cover K {k0..k0+7, k0+16..k0+23},
// lanes>=16 cover K {k0+8..k0+15, k0+24..k0+31} (pairs per VGPR).
__device__ __forceinline__ v16bf load_a_tile(const unsigned short* __restrict__ plane,
                                             int row, int ldk, int k0, int lane) {
    const int half = (lane < 16) ? 0 : 8;
    const unsigned short* p = plane + (size_t)row * ldk + k0 + half;
    union { uint4 q[2]; v16bf v; } u;
    u.q[0] = *(const uint4*)(p);
    u.q[1] = *(const uint4*)(p + 16);
    return u.v;
}

// B-matrix 32x16 bf16 lane layout: lane n=L%16; lanes<16 cover K {k0..k0+7, k0+16..k0+23},
// lanes>=16 cover K {k0+8..k0+15, k0+24..k0+31}.
__device__ __forceinline__ v16bf load_b_tile(const unsigned short* __restrict__ plane,
                                             int row, int ldk, int k0, int lane) {
    const int half = (lane < 16) ? 0 : 8;
    const unsigned short* p = plane + (size_t)row * ldk + k0 + half;
    union { uint4 q[2]; v16bf v; } u;
    u.q[0] = *(const uint4*)(p);
    u.q[1] = *(const uint4*)(p + 16);
    return u.v;
}

// ---------------------------------------------------------------------------
// Generic split-bf16 GEMM:  C(MxN) = act( A(MxK) * B(NxK)^T + bias )
// A, B given as hi/lo bf16 planes; product = hihi + hilo + lohi (f32-grade).
// Shapes are compile-time; one wave owns one full 16(M) x 16*NSUBS(N) tile.
// Flat wave-tile mapping => straight-line control flow, EXEC always all-1s.
// Launch: total_waves = (M/16) * (N/(16*NSUBS)); block = 256 (8 waves).
// ---------------------------------------------------------------------------
template <int N, int K, int NSUBS, bool RELU, bool STORE_BF16>
__global__ __launch_bounds__(256)
void made_wmma_gemm(const unsigned short* __restrict__ Ahi,
                    const unsigned short* __restrict__ Alo,
                    const unsigned short* __restrict__ Bhi,
                    const unsigned short* __restrict__ Blo,
                    const float* __restrict__ bias,
                    float* __restrict__ Cf,
                    unsigned short* __restrict__ Chi,
                    unsigned short* __restrict__ Clo) {
    constexpr int TILES_N = N / (16 * NSUBS);
    const int lane  = threadIdx.x & 31;
    const int gwave = blockIdx.x * 8 + (threadIdx.x >> 5);
    const int m_base = (gwave / TILES_N) * 16;
    const int n_wave = (gwave % TILES_N) * (16 * NSUBS);

    const int arow = m_base + (lane & 15);
    v8f acc[NSUBS];
#pragma unroll
    for (int s = 0; s < NSUBS; ++s) acc[s] = v8f{};

    for (int k0 = 0; k0 < K; k0 += 32) {
        const v16bf ah = load_a_tile(Ahi, arow, K, k0, lane);
        const v16bf al = load_a_tile(Alo, arow, K, k0, lane);
#pragma unroll
        for (int s = 0; s < NSUBS; ++s) {
            const int brow = n_wave + s * 16 + (lane & 15);
            const v16bf bh = load_b_tile(Bhi, brow, K, k0, lane);
            const v16bf bl = load_b_tile(Blo, brow, K, k0, lane);
            acc[s] = wmma_bf16(ah, bh, acc[s]);
            acc[s] = wmma_bf16(ah, bl, acc[s]);
            acc[s] = wmma_bf16(al, bh, acc[s]);
        }
    }

    const int rbase = m_base + ((lane >= 16) ? 8 : 0);
#pragma unroll
    for (int s = 0; s < NSUBS; ++s) {
        const int col = n_wave + s * 16 + (lane & 15);
        const float bv = bias[col];
#pragma unroll
        for (int r = 0; r < 8; ++r) {
            float v = acc[s][r] + bv;
            if (RELU) v = fmaxf(v, 0.0f);
            const size_t idx = (size_t)(rbase + r) * N + col;
            if (STORE_BF16) {
                unsigned short hi = f2bf(v);
                Chi[idx] = hi;
                Clo[idx] = f2bf(v - bf2f(hi));
            } else {
                Cf[idx] = v;
            }
        }
    }
}

// ---------------------------------------------------------------------------
// Weight preparation (masks + hi/lo split). Run once.
// m1[j,k] = (j%15 >= k);  m2[j,k] = (j%15 >= k%15);  m3[o,k] = (o%16-1 >= k%15)
// ---------------------------------------------------------------------------
__global__ void made_prep_w1(const float* __restrict__ W1, float* __restrict__ W1m) {
    int idx = blockIdx.x * blockDim.x + threadIdx.x;
    if (idx >= NH * NIN) return;
    int j = idx >> 4, k = idx & 15;
    W1m[idx] = ((j % 15) >= k) ? W1[idx] : 0.0f;
}

__global__ void made_prep_w2(const float* __restrict__ W2,
                             unsigned short* __restrict__ hi,
                             unsigned short* __restrict__ lo) {
    int idx = blockIdx.x * blockDim.x + threadIdx.x;
    if (idx >= NH * NH) return;
    int j = idx >> 10, k = idx & 1023;
    float v = ((j % 15) >= (k % 15)) ? W2[idx] : 0.0f;
    unsigned short h = f2bf(v);
    hi[idx] = h;
    lo[idx] = f2bf(v - bf2f(h));
}

__global__ void made_prep_w3(const float* __restrict__ W3,
                             unsigned short* __restrict__ hi,
                             unsigned short* __restrict__ lo) {
    int idx = blockIdx.x * blockDim.x + threadIdx.x;
    if (idx >= 2 * NIN * NH) return;
    int o = idx >> 10, k = idx & 1023;
    float v = (((o & 15) - 1) >= (k % 15)) ? W3[idx] : 0.0f;
    unsigned short h = f2bf(v);
    hi[idx] = h;
    lo[idx] = f2bf(v - bf2f(h));
}

// hc[b,j] = cond[b,:] . Wc[j,:] + b1[j]   (loop-invariant, exact f32)
__global__ __launch_bounds__(256)
void made_prep_hc(const float* __restrict__ cond, const float* __restrict__ Wc,
                  const float* __restrict__ b1, float* __restrict__ hc) {
    int j = blockIdx.x * blockDim.x + threadIdx.x;   // 0..1023
    int b = blockIdx.y;
    const float4* c4 = (const float4*)(cond + (size_t)b * NCOND);
    const float4* w4 = (const float4*)(Wc + (size_t)j * NCOND);
    float s = b1[j];
#pragma unroll 8
    for (int t = 0; t < NCOND / 4; ++t) {
        float4 c = c4[t], w = w4[t];
        s += c.x * w.x + c.y * w.y + c.z * w.z + c.w * w.w;
    }
    hc[(size_t)b * NH + j] = s;
}

__global__ void made_init_x(float* __restrict__ x) {
    int idx = blockIdx.x * blockDim.x + threadIdx.x;
    if (idx < BATCH * NIN) x[idx] = 0.0f;
}

// ---------------------------------------------------------------------------
// Joiner: h1[b,j] = relu( x[b,:16] . W1m[j,:16] + hc[b,j] ) -> bf16 hi/lo planes
// ---------------------------------------------------------------------------
__global__ __launch_bounds__(256)
void made_joiner(const float* __restrict__ x, const float* __restrict__ W1m,
                 const float* __restrict__ hc,
                 unsigned short* __restrict__ Hhi, unsigned short* __restrict__ Hlo) {
    int j = blockIdx.x * blockDim.x + threadIdx.x;   // 0..1023
    int b = blockIdx.y;
    const float4* xr = (const float4*)(x + (size_t)b * NIN);
    const float4* wr = (const float4*)(W1m + (size_t)j * NIN);
    float s = hc[(size_t)b * NH + j];
#pragma unroll
    for (int t = 0; t < NIN / 4; ++t) {
        float4 xv = xr[t], wv = wr[t];
        s += xv.x * wv.x + xv.y * wv.y + xv.z * wv.z + xv.w * wv.w;
    }
    s = fmaxf(s, 0.0f);
    unsigned short h = f2bf(s);
    size_t idx = (size_t)b * NH + j;
    Hhi[idx] = h;
    Hlo[idx] = f2bf(s - bf2f(h));
}

// ---------------------------------------------------------------------------
// Autoregressive column update; on the final step also emit d_out.
// out layout: (2048 x 32), m = out[:, :16], a = out[:, 16:]
// ---------------------------------------------------------------------------
__global__ void made_update(const float* __restrict__ out, const float* __restrict__ input,
                            float* __restrict__ x, float* __restrict__ dout,
                            int i, int is_final) {
    int b = blockIdx.x * blockDim.x + threadIdx.x;
    if (b >= BATCH) return;
    const float* o = out + (size_t)b * (2 * NIN);
    float m = o[i];
    float a = o[NIN + i];
    x[(size_t)b * NIN + i] = input[(size_t)b * NIN + i] * expf(a) + m;
    if (is_final) {
        float s = 0.0f;
#pragma unroll
        for (int c = 0; c < NIN; ++c) {
            dout[(size_t)b * NIN + c] = x[(size_t)b * NIN + c];
            s += o[NIN + c];
        }
        dout[(size_t)BATCH * NIN + b] = s;
    }
}

// ---------------------------------------------------------------------------
// Host-side orchestration
// ---------------------------------------------------------------------------
extern "C" void kernel_launch(void* const* d_in, const int* in_sizes, int n_in,
                              void* d_out, int out_size, void* d_ws, size_t ws_size,
                              hipStream_t stream) {
    const float* input = (const float*)d_in[0];
    const float* cond  = (const float*)d_in[1];
    const float* W1    = (const float*)d_in[2];
    const float* b1    = (const float*)d_in[3];
    const float* Wc    = (const float*)d_in[4];
    const float* W2    = (const float*)d_in[5];
    const float* b2    = (const float*)d_in[6];
    const float* W3    = (const float*)d_in[7];
    const float* b3    = (const float*)d_in[8];
    float* dout = (float*)d_out;

    // Carve workspace (256B-aligned slices)
    char* p = (char*)d_ws;
    auto carve = [&](size_t bytes) -> void* {
        void* r = (void*)p;
        p += (bytes + 255) & ~(size_t)255;
        return r;
    };
    float*          W1m  = (float*)         carve(sizeof(float) * NH * NIN);
    unsigned short* W2hi = (unsigned short*)carve(sizeof(short) * NH * NH);
    unsigned short* W2lo = (unsigned short*)carve(sizeof(short) * NH * NH);
    unsigned short* W3hi = (unsigned short*)carve(sizeof(short) * 2 * NIN * NH);
    unsigned short* W3lo = (unsigned short*)carve(sizeof(short) * 2 * NIN * NH);
    float*          hc   = (float*)         carve(sizeof(float) * (size_t)BATCH * NH);
    unsigned short* H1hi = (unsigned short*)carve(sizeof(short) * (size_t)BATCH * NH);
    unsigned short* H1lo = (unsigned short*)carve(sizeof(short) * (size_t)BATCH * NH);
    unsigned short* H2hi = (unsigned short*)carve(sizeof(short) * (size_t)BATCH * NH);
    unsigned short* H2lo = (unsigned short*)carve(sizeof(short) * (size_t)BATCH * NH);
    float*          outb = (float*)         carve(sizeof(float) * (size_t)BATCH * 2 * NIN);
    float*          xbuf = (float*)         carve(sizeof(float) * (size_t)BATCH * NIN);

    // One-time preparation (loop-invariant)
    made_prep_w1<<<(NH * NIN + 255) / 256, 256, 0, stream>>>(W1, W1m);
    made_prep_w2<<<(NH * NH + 255) / 256, 256, 0, stream>>>(W2, W2hi, W2lo);
    made_prep_w3<<<(2 * NIN * NH + 255) / 256, 256, 0, stream>>>(W3, W3hi, W3lo);
    made_prep_hc<<<dim3(NH / 256, BATCH), 256, 0, stream>>>(cond, Wc, b1, hc);
    made_init_x<<<(BATCH * NIN + 255) / 256, 256, 0, stream>>>(xbuf);

    // GEMM2: M=2048, N=1024, NSUBS=4 -> tiles = 128*16 = 2048 waves -> 256 blocks
    constexpr int GEMM2_BLOCKS = (BATCH / 16) * (NH / 64) / 8;        // 256
    // GEMM3: M=2048, N=32, NSUBS=2 -> tiles = 128*1 = 128 waves -> 16 blocks
    constexpr int GEMM3_BLOCKS = (BATCH / 16) * ((2 * NIN) / 32) / 8; // 16

    for (int i = 0; i < NIN; ++i) {
        // Layer 1: h1 = relu(x @ W1m^T + hc)  (K=16, exact f32 VALU)
        made_joiner<<<dim3(NH / 256, BATCH), 256, 0, stream>>>(xbuf, W1m, hc, H1hi, H1lo);
        // Layer 2: h2 = relu(h1 @ W2m^T + b2)  (WMMA bf16x3, dominant cost)
        made_wmma_gemm<NH, NH, 4, true, true><<<GEMM2_BLOCKS, 256, 0, stream>>>(
            H1hi, H1lo, W2hi, W2lo, b2, nullptr, H2hi, H2lo);
        // Layer 3: out = h2 @ W3m^T + b3  (WMMA bf16x3, N=32)
        made_wmma_gemm<2 * NIN, NH, 2, false, false><<<GEMM3_BLOCKS, 256, 0, stream>>>(
            H2hi, H2lo, W3hi, W3lo, b3, outb, nullptr, nullptr);
        // Autoregressive update; emit outputs on the last step
        made_update<<<(BATCH + 255) / 256, 256, 0, stream>>>(
            outb, input, xbuf, dout, i, (i == NIN - 1) ? 1 : 0);
    }
}
